// GlobalMeshRefiner_36017595744362
// MI455X (gfx1250) — compile-verified, matrix operand order checked by reading, not run on hardware
//
#include <hip/hip_runtime.h>
#include <hip/hip_bf16.h>
#include <stdint.h>

// Problem constants (from reference)
#define B_   64
#define N_   778
#define NPAD 800     // padded node count: 25 * 32 key-steps, 50 * 16 row tiles
#define C_   256
#define CQK  32
#define SP   9
#define OC   320     // 32 (q) + 32 (k) + 256 (v) output columns of fused QKV GEMM
#define AW   5       // waves per attention workgroup (5 Q tiles share K/V staging)

typedef __attribute__((ext_vector_type(16))) __bf16 v16bf;
typedef __attribute__((ext_vector_type(8)))  float  v8f;
typedef __attribute__((ext_vector_type(4)))  int    v4i;

union ABu { v16bf v; uint32_t u[8]; uint16_t s[16]; };

__device__ __forceinline__ uint16_t f2bf(float f) {
  uint32_t u = __builtin_bit_cast(uint32_t, f);
  u += 0x7FFFu + ((u >> 16) & 1u);          // round-to-nearest-even
  return (uint16_t)(u >> 16);
}

#define GAS __attribute__((address_space(1)))
#define LAS __attribute__((address_space(3)))

// 16-byte global -> LDS copy. Async (tracked by ASYNCcnt) when the gfx1250
// builtin is available; synchronous fallback otherwise.
__device__ __forceinline__ void copy16_g2l(const uint16_t* gsrc, uint16_t* ldst) {
#if __has_builtin(__builtin_amdgcn_global_load_async_to_lds_b128)
  __builtin_amdgcn_global_load_async_to_lds_b128(
      (GAS v4i*)gsrc, (LAS v4i*)ldst, 0, 0);
#else
  *(uint4*)ldst = *(const uint4*)gsrc;
#endif
}

__device__ __forceinline__ void copy16_g2l_wait() {
#if __has_builtin(__builtin_amdgcn_global_load_async_to_lds_b128) && \
    __has_builtin(__builtin_amdgcn_s_wait_asynccnt)
  __builtin_amdgcn_s_wait_asynccnt(0);
#endif
}

// ---------------------------------------------------------------------------
// Kernel 1: x = concat(feature, coord2d) -> bf16  [B*N, 256]
// ---------------------------------------------------------------------------
__global__ void k_prep_x(const float* __restrict__ feat,
                         const float* __restrict__ c2d,
                         uint16_t* __restrict__ xbf) {
  int idx = blockIdx.x * 256 + threadIdx.x;
  if (idx >= B_ * N_ * C_) return;
  int c  = idx & 255;
  int bn = idx >> 8;
  float v = (c < 254) ? feat[(size_t)bn * 254 + c]
                      : c2d[(size_t)bn * 2 + (c - 254)];
  xbf[idx] = f2bf(v);
}

// ---------------------------------------------------------------------------
// Kernel 2: fused transposed weight matrix wT[320][256] in bf16
// ---------------------------------------------------------------------------
__global__ void k_prep_w(const float* __restrict__ Wq,
                         const float* __restrict__ Wk,
                         const float* __restrict__ Wv,
                         uint16_t* __restrict__ wT) {
  int idx = blockIdx.x * 256 + threadIdx.x;
  if (idx >= OC * C_) return;
  int oc = idx >> 8;       // output column
  int k  = idx & 255;      // input channel
  float v;
  if (oc < 32)       v = Wq[(size_t)k * 32  + oc];
  else if (oc < 64)  v = Wk[(size_t)k * 32  + (oc - 32)];
  else               v = Wv[(size_t)k * 256 + (oc - 64)];
  wT[idx] = f2bf(v);
}

// ---------------------------------------------------------------------------
// Kernel 3: zero pad rows (n in [778,800)) of q, k and pad keys of vT
// ---------------------------------------------------------------------------
__global__ void k_zero_pad(uint16_t* __restrict__ q,
                           uint16_t* __restrict__ k,
                           uint16_t* __restrict__ vT) {
  const int PADN = NPAD - N_;                 // 22
  const int QP   = B_ * PADN * CQK;           // 45056
  const int VP   = B_ * C_ * PADN;            // 360448
  int idx = blockIdx.x * 256 + threadIdx.x;
  if (idx < QP) {
    int b = idx / (PADN * CQK);
    int r = idx % (PADN * CQK);
    int n = N_ + r / CQK;
    int c = r % CQK;
    q[((size_t)b * NPAD + n) * CQK + c] = 0;
  } else if (idx < 2 * QP) {
    int j = idx - QP;
    int b = j / (PADN * CQK);
    int r = j % (PADN * CQK);
    int n = N_ + r / CQK;
    int c = r % CQK;
    k[((size_t)b * NPAD + n) * CQK + c] = 0;
  } else if (idx < 2 * QP + VP) {
    int j  = idx - 2 * QP;
    int b  = j / (C_ * PADN);
    int r  = j % (C_ * PADN);
    int ch = r / PADN;
    int n  = N_ + r % PADN;
    vT[((size_t)b * C_ + ch) * NPAD + n] = 0;
  }
}

// ---------------------------------------------------------------------------
// Kernel 4: fused QKV GEMM via WMMA.
//   [B*N, 256] x [256, 320] -> q/k bf16 row-major, v transposed (vT[b][ch][n])
// ---------------------------------------------------------------------------
__global__ __launch_bounds__(32)
void k_qkv(const uint16_t* __restrict__ xbf, const uint16_t* __restrict__ wT,
           const float* __restrict__ bq, const float* __restrict__ bk,
           const float* __restrict__ bv,
           uint16_t* __restrict__ q, uint16_t* __restrict__ k,
           uint16_t* __restrict__ vT) {
  int lane  = threadIdx.x;
  int col   = lane & 15;
  int khalf = lane >> 4;
  int row0  = blockIdx.x * 16;
  int ct    = blockIdx.y;
  int ocol  = ct * 16 + col;

  v8f acc = {};
  const uint16_t* arow = xbf + (size_t)(row0 + col) * C_;  // A: row = lane%16
  const uint16_t* brow = wT  + (size_t)ocol * C_;          // B: col = lane%16

  for (int kk = 0; kk < 8; ++kk) {
    ABu a, b;
    int base = kk * 32 + khalf * 8;
#pragma unroll
    for (int r = 0; r < 4; ++r) {
      a.u[r]     = *(const uint32_t*)(arow + base + 2 * r);
      a.u[r + 4] = *(const uint32_t*)(arow + base + 16 + 2 * r);
      b.u[r]     = *(const uint32_t*)(brow + base + 2 * r);
      b.u[r + 4] = *(const uint32_t*)(brow + base + 16 + 2 * r);
    }
    acc = __builtin_amdgcn_wmma_f32_16x16x32_bf16(false, a.v, false, b.v,
                                                  (short)0, acc, false, false);
  }

  float bias = (ocol < 32) ? bq[ocol] : (ocol < 64 ? bk[ocol - 32] : bv[ocol - 64]);
#pragma unroll
  for (int r = 0; r < 8; ++r) {
    int rowM = r + 8 * khalf;             // C layout: vgpr r -> M=r / M=r+8
    int fr   = row0 + rowM;
    int b_   = fr / N_;
    int n_   = fr % N_;
    uint16_t val = f2bf(acc[r] + bias);
    if (ocol < 32)
      q[((size_t)b_ * NPAD + n_) * CQK + ocol] = val;
    else if (ocol < 64)
      k[((size_t)b_ * NPAD + n_) * CQK + (ocol - 32)] = val;
    else
      vT[((size_t)b_ * C_ + (ocol - 64)) * NPAD + n_] = val;
  }
}

// ---------------------------------------------------------------------------
// Kernel 5: flash attention + residual.
//   Workgroup = 5 waves, each owning one 16-row Q tile; all waves share
//   async-staged K (2 KB) and V (16 KB) tiles in LDS per 32-key step.
// ---------------------------------------------------------------------------
#define LSTRIDE 40   // LDS row stride in ushorts: 80 B, keeps 16 B alignment

__global__ __launch_bounds__(32 * AW)
void k_attn(const uint16_t* __restrict__ q, const uint16_t* __restrict__ kmat,
            const uint16_t* __restrict__ vT,
            const float* __restrict__ feat, const float* __restrict__ c2d,
            const float* __restrict__ gamma_p, float* __restrict__ att) {
  __shared__ uint16_t smk[32 * LSTRIDE];            // K tile: 32 keys x 32 ch
  __shared__ uint16_t smv[256 * LSTRIDE];           // V tile: 256 ch x 32 keys
  __shared__ uint16_t pl[AW][16][32];               // P transpose staging

  int tid   = threadIdx.x;
  int lane  = tid & 31;
  int wave  = tid >> 5;
  int col   = lane & 15;
  int khalf = (lane >> 4) & 1;
  int tile  = blockIdx.x * AW + wave;               // 0..49
  int b     = blockIdx.y;
  float gamma = gamma_p[0];

  // Q tile, bf16 A layout (per-wave, from global)
  ABu qa;
  {
    const uint16_t* qrow =
        q + ((size_t)b * NPAD + tile * 16 + col) * CQK + khalf * 8;
#pragma unroll
    for (int r = 0; r < 4; ++r) {
      qa.u[r]     = *(const uint32_t*)(qrow + 2 * r);
      qa.u[r + 4] = *(const uint32_t*)(qrow + 16 + 2 * r);
    }
  }

  float m[8], l[8];
#pragma unroll
  for (int r = 0; r < 8; ++r) { m[r] = -3.0e38f; l[r] = 0.0f; }
  v8f acc[16];
  const v8f vzero = {};
#pragma unroll
  for (int c = 0; c < 16; ++c) acc[c] = vzero;

  const uint16_t* kbase = kmat + (size_t)b * NPAD * CQK;
  const uint16_t* vbase = vT + (size_t)b * C_ * NPAD;

  for (int j = 0; j < 25; ++j) {
    int key0 = j * 32;

    // ---- cooperative async staging of K (128 x 16B) and V (1024 x 16B) ----
    __syncthreads();                       // previous step's consumers done
    for (int i = tid; i < 128 + 1024; i += 32 * AW) {
      if (i < 128) {
        int row = i >> 2, part = i & 3;
        copy16_g2l(kbase + (size_t)(key0 + row) * CQK + part * 8,
                   &smk[row * LSTRIDE + part * 8]);
      } else {
        int t = i - 128;
        int ch = t >> 2, part = t & 3;
        copy16_g2l(vbase + (size_t)ch * NPAD + key0 + part * 8,
                   &smv[ch * LSTRIDE + part * 8]);
      }
    }
    copy16_g2l_wait();
    __syncthreads();

    // ---- scores: S = Q @ K^T (two 16-key tiles) ----
    ABu k1, k2;
#pragma unroll
    for (int r = 0; r < 4; ++r) {
      int ck = khalf * 8 + 2 * r;
      k1.u[r]     = *(const uint32_t*)&smk[col * LSTRIDE + ck];
      k1.u[r + 4] = *(const uint32_t*)&smk[col * LSTRIDE + 16 + ck];
      k2.u[r]     = *(const uint32_t*)&smk[(16 + col) * LSTRIDE + ck];
      k2.u[r + 4] = *(const uint32_t*)&smk[(16 + col) * LSTRIDE + 16 + ck];
    }
    v8f s1 = vzero, s2 = vzero;
    s1 = __builtin_amdgcn_wmma_f32_16x16x32_bf16(false, qa.v, false, k1.v,
                                                 (short)0, s1, false, false);
    s2 = __builtin_amdgcn_wmma_f32_16x16x32_bf16(false, qa.v, false, k2.v,
                                                 (short)0, s2, false, false);

    bool mask1 = (key0 + col) >= N_;
    bool mask2 = (key0 + 16 + col) >= N_;
    float p1[8], p2[8], scale[8];
#pragma unroll
    for (int r = 0; r < 8; ++r) {
      float a1 = mask1 ? -1.0e30f : s1[r];
      float a2 = mask2 ? -1.0e30f : s2[r];
      float t = fmaxf(a1, a2);
#pragma unroll
      for (int off = 1; off < 16; off <<= 1)
        t = fmaxf(t, __shfl_xor(t, off, 32));   // 16-lane row-group reduce
      float mn = fmaxf(m[r], t);
      float sc = __expf(m[r] - mn);
      float e1 = __expf(a1 - mn);
      float e2 = __expf(a2 - mn);
      float rs = e1 + e2;
#pragma unroll
      for (int off = 1; off < 16; off <<= 1)
        rs += __shfl_xor(rs, off, 32);
      l[r] = l[r] * sc + rs;
      m[r] = mn;
      p1[r] = e1; p2[r] = e2; scale[r] = sc;
    }
#pragma unroll
    for (int c = 0; c < 16; ++c)
#pragma unroll
      for (int r = 0; r < 8; ++r) acc[c][r] *= scale[r];

    // ---- C-layout probabilities -> bf16 A layout via per-wave LDS tile ----
#pragma unroll
    for (int r = 0; r < 8; ++r) {
      int rowM = r + 8 * khalf;
      pl[wave][rowM][col]      = f2bf(p1[r]);
      pl[wave][rowM][16 + col] = f2bf(p2[r]);
    }
    __syncthreads();
    ABu pa;
#pragma unroll
    for (int r = 0; r < 4; ++r) {
      pa.u[r]     = *(const uint32_t*)&pl[wave][col][khalf * 8 + 2 * r];
      pa.u[r + 4] = *(const uint32_t*)&pl[wave][col][16 + khalf * 8 + 2 * r];
    }
    __syncthreads();

    // ---- O += P @ V over 16 channel chunks, V operand from LDS ----
#pragma unroll
    for (int c = 0; c < 16; ++c) {
      ABu vb;
      int ch = c * 16 + col;
#pragma unroll
      for (int r = 0; r < 4; ++r) {
        int kl = khalf * 8 + 2 * r;
        vb.u[r]     = *(const uint32_t*)&smv[ch * LSTRIDE + kl];
        vb.u[r + 4] = *(const uint32_t*)&smv[ch * LSTRIDE + 16 + kl];
      }
      acc[c] = __builtin_amdgcn_wmma_f32_16x16x32_bf16(
          false, pa.v, false, vb.v, (short)0, acc[c], false, false);
    }
  }

  // epilogue: normalize, residual, store f32
#pragma unroll
  for (int r = 0; r < 8; ++r) {
    int rowM = r + 8 * khalf;
    int n = tile * 16 + rowM;
    if (n >= N_) continue;
    float invl = 1.0f / l[r];
    size_t bn = (size_t)b * N_ + n;
#pragma unroll
    for (int c = 0; c < 16; ++c) {
      int ch = c * 16 + col;
      float x = (ch < 254) ? feat[bn * 254 + ch] : c2d[bn * 2 + (ch - 254)];
      att[bn * C_ + ch] = gamma * acc[c][r] * invl + x;
    }
  }
}

// ---------------------------------------------------------------------------
// Kernel 6: SpiralConv head. One wave per (b, n); 2304-long dot x 3 outputs.
// ---------------------------------------------------------------------------
__global__ __launch_bounds__(32)
void k_head(const float* __restrict__ att, const int* __restrict__ indices,
            const float* __restrict__ Ws, const float* __restrict__ bs,
            const float* __restrict__ c3d, float* __restrict__ out) {
  int lane = threadIdx.x;
  int bn = blockIdx.x;
  int b = bn / N_, n = bn % N_;
  float a0 = 0.f, a1 = 0.f, a2 = 0.f;
  for (int e = lane; e < SP * C_; e += 32) {
    int s = e >> 8;
    int c = e & 255;
    int idx = indices[n * SP + s];
    float g = att[((size_t)b * N_ + idx) * C_ + c];
    a0 += g * Ws[(size_t)e * 3 + 0];
    a1 += g * Ws[(size_t)e * 3 + 1];
    a2 += g * Ws[(size_t)e * 3 + 2];
  }
#pragma unroll
  for (int off = 1; off < 32; off <<= 1) {
    a0 += __shfl_xor(a0, off, 32);
    a1 += __shfl_xor(a1, off, 32);
    a2 += __shfl_xor(a2, off, 32);
  }
  if (lane == 0) {
    size_t o = (size_t)bn * 3;
    out[o + 0] = 0.5f * (a0 + bs[0]) + c3d[o + 0];
    out[o + 1] = 0.5f * (a1 + bs[1]) + c3d[o + 1];
    out[o + 2] = 0.5f * (a2 + bs[2]) + c3d[o + 2];
  }
}

// ---------------------------------------------------------------------------
extern "C" void kernel_launch(void* const* d_in, const int* in_sizes, int n_in,
                              void* d_out, int out_size, void* d_ws, size_t ws_size,
                              hipStream_t stream) {
  const float* feat = (const float*)d_in[0];
  const float* c2d  = (const float*)d_in[1];
  const float* c3d  = (const float*)d_in[2];
  const int*   idx  = (const int*)d_in[3];
  const float* Wq   = (const float*)d_in[4];
  const float* bq   = (const float*)d_in[5];
  const float* Wk   = (const float*)d_in[6];
  const float* bk   = (const float*)d_in[7];
  const float* Wv   = (const float*)d_in[8];
  const float* bv   = (const float*)d_in[9];
  const float* gam  = (const float*)d_in[10];
  const float* Ws   = (const float*)d_in[11];
  const float* bs_  = (const float*)d_in[12];

  char* ws = (char*)d_ws;
  size_t off = 0;
  auto alloc = [&](size_t bytes) {
    void* p = ws + off;
    off = (off + bytes + 255) & ~(size_t)255;
    return p;
  };
  uint16_t* xbf = (uint16_t*)alloc((size_t)B_ * N_ * C_ * 2);       // ~25.5 MB
  uint16_t* wT  = (uint16_t*)alloc((size_t)OC * C_ * 2);            // 160 KB
  uint16_t* qb  = (uint16_t*)alloc((size_t)B_ * NPAD * CQK * 2);    // 3.3 MB
  uint16_t* kb  = (uint16_t*)alloc((size_t)B_ * NPAD * CQK * 2);    // 3.3 MB
  uint16_t* vT  = (uint16_t*)alloc((size_t)B_ * C_ * NPAD * 2);     // 26.2 MB
  float*    att = (float*)alloc((size_t)B_ * N_ * C_ * 4);          // 51 MB

  k_prep_x<<<(B_ * N_ * C_ + 255) / 256, 256, 0, stream>>>(feat, c2d, xbf);
  k_prep_w<<<(OC * C_ + 255) / 256, 256, 0, stream>>>(Wq, Wk, Wv, wT);
  {
    int padtotal = 2 * B_ * (NPAD - N_) * CQK + B_ * C_ * (NPAD - N_);
    k_zero_pad<<<(padtotal + 255) / 256, 256, 0, stream>>>(qb, kb, vT);
  }
  {
    dim3 g((B_ * N_) / 16, OC / 16);     // 3112 x 20
    k_qkv<<<g, 32, 0, stream>>>(xbf, wT, bq, bk, bv, qb, kb, vT);
  }
  {
    dim3 g((NPAD / 16) / AW, B_);        // 10 x 64, 5 waves each
    k_attn<<<g, 32 * AW, 0, stream>>>(qb, kb, vT, feat, c2d, gam, att);
  }
  k_head<<<B_ * N_, 32, 0, stream>>>(att, idx, Ws, bs_, c3d, (float*)d_out);
}
